// TractCloud_56908316672770
// MI455X (gfx1250) — matrix operand order repaired and checked by reading, not compiled, not run on hardware
//
#include <hip/hip_runtime.h>
#include <hip/hip_bf16.h>

typedef __attribute__((ext_vector_type(16))) _Float16 v16h;
typedef __attribute__((ext_vector_type(8)))  float    v8f;

#define BB  256
#define NN  15
#define KK  520
#define CLS 800
#define EPSBN 1e-5f

struct alignas(16) H8 { _Float16 h[8]; };

// single-instruction max (skip llvm.maxnum's canonicalize of both inputs;
// WMMA outputs are already canonical per CDNA5 RNE/denorm-preserve rules)
__device__ inline float fmax_raw(float a, float b) {
  float r;
  asm("v_max_num_f32 %0, %1, %2" : "=v"(r) : "v"(a), "v"(b));
  return r;
}

// ---------------- fragment loaders (CDNA5 WMMA 16x16x32 f16 layouts) -------
// A (16x32, MxK): lane m=L&15; lanes 0-15 hold K {0..7,16..23}, lanes 16-31 hold {8..15,24..31}
__device__ inline v16h load_a_frag(const _Float16* A, int ld, int row0, int k0, int lane) {
  int m  = lane & 15;
  int kb = k0 + ((lane >> 4) << 3);              // +0 or +8
  const _Float16* p = A + (size_t)(row0 + m) * ld + kb;
  H8 lo = *(const H8*)p;
  H8 hi = *(const H8*)(p + 16);
  v16h a;
#pragma unroll
  for (int i = 0; i < 8; ++i) { a[i] = lo.h[i]; a[8 + i] = hi.h[i]; }
  return a;
}

// B (32x16, KxN): lane n=L&15 is the column; lanes 0-15 hold K {0..7,16..23}, lanes 16-31 hold {8..15,24..31}
// W is row-major [Nout, K] so a column's K-run is contiguous.
__device__ inline v16h load_b_frag(const _Float16* W, int ld, int col0, int k0, int lane) {
  int n  = lane & 15;
  int kb = k0 + ((lane >> 4) << 3);              // +0 or +8
  const _Float16* p = W + (size_t)(col0 + n) * ld + kb;
  H8 lo = *(const H8*)p;
  H8 hi = *(const H8*)(p + 16);
  v16h b;
#pragma unroll
  for (int i = 0; i < 8; ++i) { b[i] = lo.h[i]; b[8 + i] = hi.h[i]; }
  return b;
}

// ---------------- BN folding: W' = s*W, b' = s*(b-m)+beta, s = g*rsqrt(v+eps)
__global__ void fold_bn_kernel(const float* __restrict__ W, const float* __restrict__ b,
                               const float* __restrict__ g, const float* __restrict__ beta,
                               const float* __restrict__ mu, const float* __restrict__ var,
                               _Float16* __restrict__ Wh, float* __restrict__ bf,
                               int O, int C) {
  int t = blockIdx.x * blockDim.x + threadIdx.x;
  if (t < O * C) {
    int o = t / C;
    float s = g[o] * rsqrtf(var[o] + EPSBN);
    Wh[t] = (_Float16)(s * W[t]);
  }
  if (t < O) {
    float s = g[t] * rsqrtf(var[t] + EPSBN);
    bf[t] = s * (b[t] - mu[t]) + beta[t];
  }
}

__global__ void to_half_kernel(const float* __restrict__ W, _Float16* __restrict__ Wh, int n) {
  int t = blockIdx.x * blockDim.x + threadIdx.x;
  if (t < n) Wh[t] = (_Float16)W[t];
}

// ---------------- edge MLP (6->64) + max over K neighbors, one wave per (b,n)
// A = folded W1 padded to 16x32 tiles (4 channel tiles), B = 16-neighbor feature tiles.
__global__ void edge_kernel(const float* __restrict__ pts, const float* __restrict__ ips,
                            const _Float16* __restrict__ W1, const float* __restrict__ b1,
                            _Float16* __restrict__ h1) {
  int wave = (int)((blockIdx.x * blockDim.x + threadIdx.x) >> 5);
  int lane = threadIdx.x & 31;
  if (wave >= BB * NN) return;
  int b = wave / NN, n = wave % NN;
  const float* xp = pts + ((size_t)b * NN + n) * 3;
  float x0 = xp[0], x1 = xp[1], x2 = xp[2];

  int m16 = lane & 15;
  v16h afr[4];
#pragma unroll
  for (int t = 0; t < 4; ++t) {
    v16h a = {};                              // pad contraction 6 -> 32 with zeros
    if (lane < 16) {                          // lanes>=16 carry K 8..15/24..31: all zero
      const _Float16* w = W1 + (t * 16 + m16) * 6;
      a[0] = w[0]; a[1] = w[1]; a[2] = w[2]; a[3] = w[3]; a[4] = w[4]; a[5] = w[5];
    }
    afr[t] = a;
  }

  v8f acc[4];
#pragma unroll
  for (int t = 0; t < 4; ++t)
#pragma unroll
    for (int r = 0; r < 8; ++r) acc[t][r] = -3.4e38f;

  for (int ci = 0; ci < (KK + 15) / 16; ++ci) {
    v16h bf = {};
    if (lane < 16) {                          // one neighbor per lane (column of B)
      int k = ci * 16 + m16; if (k >= KK) k = KK - 1;   // duplicate: neutral for max
      const float* ep = ips + (((size_t)b * KK + k) * NN + n) * 3;
      if (k + 16 < KK) __builtin_prefetch(ep + 16 * NN * 3, 0, 0);  // next chunk (HBM stream)
      bf[0] = (_Float16)(ep[0] - x0);
      bf[1] = (_Float16)(ep[1] - x1);
      bf[2] = (_Float16)(ep[2] - x2);
      bf[3] = (_Float16)x0;
      bf[4] = (_Float16)x1;
      bf[5] = (_Float16)x2;
    }
    // EXEC is all-ones again here; 4 channel tiles x one neighbor tile
#pragma unroll
    for (int t = 0; t < 4; ++t) {
      v8f z = {};
      v8f d = __builtin_amdgcn_wmma_f32_16x16x32_f16(false, afr[t], false, bf,
                                                     (short)0, z, false, false);
#pragma unroll
      for (int r = 0; r < 8; ++r) acc[t][r] = fmax_raw(acc[t][r], d[r]);
    }
  }

  // reduce max across the 16 neighbor columns (lanes within each half)
#pragma unroll
  for (int t = 0; t < 4; ++t)
#pragma unroll
    for (int r = 0; r < 8; ++r) {
      float v = acc[t][r];
      v = fmax_raw(v, __shfl_xor(v, 1, 32));
      v = fmax_raw(v, __shfl_xor(v, 2, 32));
      v = fmax_raw(v, __shfl_xor(v, 4, 32));
      v = fmax_raw(v, __shfl_xor(v, 8, 32));
      acc[t][r] = v;
    }

  int hi = lane >> 4;
  if (m16 == 0) {                             // lanes 0 and 16 write ch 0..7 / 8..15 of each tile
    _Float16* o = h1 + (size_t)wave * 64;
#pragma unroll
    for (int t = 0; t < 4; ++t)
#pragma unroll
      for (int r = 0; r < 8; ++r) {
        int ch = t * 16 + r + (hi << 3);
        float v = acc[t][r] + b1[ch];
        o[ch] = (_Float16)fmaxf(v, 0.0f);     // bias+ReLU after max (commute)
      }
  }
}

// ---------------- 32x32 macro-tile WMMA GEMM: out = act(A @ W^T + bias)
// 2 A-frags x 2 B-frags -> 4 WMMAs per k-step: 2x the FLOP/byte of a 16x16 tile.
__global__ void gemm32_kernel(const _Float16* __restrict__ A, const _Float16* __restrict__ W,
                              const float* __restrict__ bias,
                              _Float16* __restrict__ outH, float* __restrict__ outF,
                              int Mrows, int Kdim, int Nout, int doRelu) {
  int wave = (int)((blockIdx.x * blockDim.x + threadIdx.x) >> 5);
  int lane = threadIdx.x & 31;
  int tilesN = Nout >> 5;
  int tilesM = Mrows >> 5;
  if (wave >= tilesM * tilesN) return;        // wave-uniform: EXEC stays all-ones
  int tm = wave / tilesN, tn = wave % tilesN;
  int row0 = tm << 5, col0 = tn << 5;

  v8f c00 = {}, c01 = {}, c10 = {}, c11 = {};
  for (int k0 = 0; k0 < Kdim; k0 += 32) {
    v16h a0 = load_a_frag(A, Kdim, row0,      k0, lane);
    v16h a1 = load_a_frag(A, Kdim, row0 + 16, k0, lane);
    v16h b0 = load_b_frag(W, Kdim, col0,      k0, lane);
    v16h b1 = load_b_frag(W, Kdim, col0 + 16, k0, lane);
    if (k0 + 32 < Kdim) {                     // gfx1250 global_prefetch_b8 of next k-slice
      __builtin_prefetch(A + (size_t)(row0 + (lane & 15)) * Kdim + k0 + 32, 0, 0);
      __builtin_prefetch(W + (size_t)(col0 + (lane & 15)) * Kdim + k0 + 32, 0, 0);
    }
    c00 = __builtin_amdgcn_wmma_f32_16x16x32_f16(false, a0, false, b0, (short)0, c00, false, false);
    c01 = __builtin_amdgcn_wmma_f32_16x16x32_f16(false, a0, false, b1, (short)0, c01, false, false);
    c10 = __builtin_amdgcn_wmma_f32_16x16x32_f16(false, a1, false, b0, (short)0, c10, false, false);
    c11 = __builtin_amdgcn_wmma_f32_16x16x32_f16(false, a1, false, b1, (short)0, c11, false, false);
  }

  int n = lane & 15, hi = lane >> 4;
  v8f* cc[2][2] = { { &c00, &c01 }, { &c10, &c11 } };
#pragma unroll
  for (int i = 0; i < 2; ++i)
#pragma unroll
    for (int j = 0; j < 2; ++j) {
      int col = col0 + j * 16 + n;
      float bv = bias ? bias[col] : 0.0f;
#pragma unroll
      for (int r = 0; r < 8; ++r) {
        float v = (*cc[i][j])[r] + bv;
        if (doRelu) v = fmaxf(v, 0.0f);
        int row = row0 + i * 16 + r + (hi << 3);
        size_t idx = (size_t)row * Nout + col;
        if (outH) outH[idx] = (_Float16)v;
        if (outF) outF[idx] = v;
      }
    }
}

// ---------------- global max-pool over N streamline points
__global__ void maxn_kernel(const float* __restrict__ h3, _Float16* __restrict__ gH) {
  int idx = blockIdx.x * blockDim.x + threadIdx.x;
  if (idx >= BB * 1024) return;
  int b = idx >> 10, ch = idx & 1023;
  const float* p = h3 + (size_t)b * NN * 1024 + ch;
  float m = p[0];
  for (int nn = 1; nn < NN; ++nn) m = fmaxf(m, p[(size_t)nn * 1024]);
  gH[idx] = (_Float16)m;
}

// ---------------------------------------------------------------------------
extern "C" void kernel_launch(void* const* d_in, const int* in_sizes, int n_in,
                              void* d_out, int out_size, void* d_ws, size_t ws_size,
                              hipStream_t stream) {
  const float* pts    = (const float*)d_in[0];
  /* d_in[1] = label (unused) */
  const float* ips    = (const float*)d_in[2];
  const float* w_info = (const float*)d_in[3];
  const float* b_info = (const float*)d_in[4];
  const float* g_info = (const float*)d_in[5];
  const float* be_info= (const float*)d_in[6];
  const float* m_info = (const float*)d_in[7];
  const float* v_info = (const float*)d_in[8];
  const float* w2  = (const float*)d_in[9];
  const float* b2  = (const float*)d_in[10];
  const float* g2  = (const float*)d_in[11];
  const float* be2 = (const float*)d_in[12];
  const float* m2  = (const float*)d_in[13];
  const float* v2  = (const float*)d_in[14];
  const float* w3  = (const float*)d_in[15];
  const float* b3  = (const float*)d_in[16];
  const float* g3  = (const float*)d_in[17];
  const float* be3 = (const float*)d_in[18];
  const float* m3  = (const float*)d_in[19];
  const float* v3  = (const float*)d_in[20];
  const float* wf1 = (const float*)d_in[21];
  const float* bf1 = (const float*)d_in[22];
  const float* gf1 = (const float*)d_in[23];
  const float* bef1= (const float*)d_in[24];
  const float* mf1 = (const float*)d_in[25];
  const float* vf1 = (const float*)d_in[26];
  const float* wf2 = (const float*)d_in[27];
  const float* bf2 = (const float*)d_in[28];
  const float* gf2 = (const float*)d_in[29];
  const float* bef2= (const float*)d_in[30];
  const float* mf2 = (const float*)d_in[31];
  const float* vf2 = (const float*)d_in[32];
  const float* wf3 = (const float*)d_in[33];
  const float* bf3 = (const float*)d_in[34];

  // ---- workspace carve-out (256B aligned)
  char* ws = (char*)d_ws;
  size_t off = 0;
  auto carve = [&](size_t bytes) -> void* {
    void* p = ws + off;
    off = (off + bytes + 255) & ~(size_t)255;
    return p;
  };
  _Float16* W1h  = (_Float16*)carve(64 * 6 * 2);
  float*    b1f  = (float*)   carve(64 * 4);
  _Float16* W2h  = (_Float16*)carve(128 * 64 * 2);
  float*    b2f  = (float*)   carve(128 * 4);
  _Float16* W3h  = (_Float16*)carve(1024 * 128 * 2);
  float*    b3f  = (float*)   carve(1024 * 4);
  _Float16* Wf1h = (_Float16*)carve(512 * 1024 * 2);
  float*    bf1f = (float*)   carve(512 * 4);
  _Float16* Wf2h = (_Float16*)carve(256 * 512 * 2);
  float*    bf2f = (float*)   carve(256 * 4);
  _Float16* Wf3h = (_Float16*)carve((size_t)CLS * 256 * 2);
  _Float16* h1   = (_Float16*)carve((size_t)BB * NN * 64 * 2);
  _Float16* h2   = (_Float16*)carve((size_t)BB * NN * 128 * 2);
  float*    h3   = (float*)   carve((size_t)BB * NN * 1024 * 4);
  _Float16* gH   = (_Float16*)carve((size_t)BB * 1024 * 2);
  _Float16* f1h  = (_Float16*)carve((size_t)BB * 512 * 2);
  _Float16* f2h  = (_Float16*)carve((size_t)BB * 256 * 2);

  float* outRet = (float*)d_out;                      // [256, 800]
  float* outF2  = (float*)d_out + (size_t)BB * CLS;   // [256, 256]

  const int T = 256;
  auto cdiv = [](int a, int b) { return (a + b - 1) / b; };

  // ---- fold BN into weights (6 tiny prep launches)
  hipLaunchKernelGGL(fold_bn_kernel, dim3(cdiv(64 * 6, T)), dim3(T), 0, stream,
                     w_info, b_info, g_info, be_info, m_info, v_info, W1h, b1f, 64, 6);
  hipLaunchKernelGGL(fold_bn_kernel, dim3(cdiv(128 * 64, T)), dim3(T), 0, stream,
                     w2, b2, g2, be2, m2, v2, W2h, b2f, 128, 64);
  hipLaunchKernelGGL(fold_bn_kernel, dim3(cdiv(1024 * 128, T)), dim3(T), 0, stream,
                     w3, b3, g3, be3, m3, v3, W3h, b3f, 1024, 128);
  hipLaunchKernelGGL(fold_bn_kernel, dim3(cdiv(512 * 1024, T)), dim3(T), 0, stream,
                     wf1, bf1, gf1, bef1, mf1, vf1, Wf1h, bf1f, 512, 1024);
  hipLaunchKernelGGL(fold_bn_kernel, dim3(cdiv(256 * 512, T)), dim3(T), 0, stream,
                     wf2, bf2, gf2, bef2, mf2, vf2, Wf2h, bf2f, 256, 512);
  hipLaunchKernelGGL(to_half_kernel, dim3(cdiv(CLS * 256, T)), dim3(T), 0, stream,
                     wf3, Wf3h, CLS * 256);

  // ---- edge MLP + neighbor max: one wave per (b,n), 3840 waves
  hipLaunchKernelGGL(edge_kernel, dim3((BB * NN) / 8), dim3(T), 0, stream,
                     pts, ips, W1h, b1f, h1);

  int M = BB * NN;  // 3840
  // conv2: [3840,64] -> [3840,128], ReLU
  hipLaunchKernelGGL(gemm32_kernel, dim3(cdiv((M / 32) * (128 / 32), 8)), dim3(T), 0, stream,
                     h1, W2h, b2f, h2, (float*)nullptr, M, 64, 128, 1);
  // conv3: [3840,128] -> [3840,1024], no ReLU (max-pool next)
  hipLaunchKernelGGL(gemm32_kernel, dim3(cdiv((M / 32) * (1024 / 32), 8)), dim3(T), 0, stream,
                     h2, W3h, b3f, (_Float16*)nullptr, h3, M, 128, 1024, 0);
  // global max over N
  hipLaunchKernelGGL(maxn_kernel, dim3(cdiv(BB * 1024, T)), dim3(T), 0, stream, h3, gH);
  // fc1: [256,1024] -> [256,512], ReLU
  hipLaunchKernelGGL(gemm32_kernel, dim3(cdiv((BB / 32) * (512 / 32), 8)), dim3(T), 0, stream,
                     gH, Wf1h, bf1f, f1h, (float*)nullptr, BB, 1024, 512, 1);
  // fc2: [256,512] -> [256,256], ReLU; f32 copy goes straight to output slot 2
  hipLaunchKernelGGL(gemm32_kernel, dim3(cdiv((BB / 32) * (256 / 32), 8)), dim3(T), 0, stream,
                     f1h, Wf2h, bf2f, f2h, outF2, BB, 512, 256, 1);
  // fc3: [256,256] -> [256,800], raw bias, no ReLU -> output slot 1
  hipLaunchKernelGGL(gemm32_kernel, dim3(cdiv((BB / 32) * (CLS / 32), 8)), dim3(T), 0, stream,
                     f2h, Wf3h, bf3, (_Float16*)nullptr, outRet, BB, 256, CLS, 0);
}
